// CrossAttentionFusion_52578989638287
// MI455X (gfx1250) — matrix operand order repaired and checked by reading, not compiled, not run on hardware
//
#include <hip/hip_runtime.h>

typedef __attribute__((ext_vector_type(16))) __bf16 v16bf;
typedef __attribute__((ext_vector_type(8)))  float  v8f;

#define T_DIM 4096
#define B_DIM 64
#define D_DIM 64
#define ROWS (T_DIM * B_DIM)   // 262144 rows of [D] = flattened [T,B,D]

// ---------- helpers ----------
__device__ __forceinline__ unsigned short f2bf(float x) {
  union { float f; unsigned u; } v; v.f = x;
  unsigned r = v.u + 0x7FFFu + ((v.u >> 16) & 1u);   // round-to-nearest-even
  return (unsigned short)(r >> 16);
}
__device__ __forceinline__ __bf16 bfc(unsigned short s) {
  return __builtin_bit_cast(__bf16, s);
}

// A fragment (16x32 bf16, MxK). lane<16: row M=lane, K in {0..15};
// lane>=16: row M=lane-16, K in {16..31}. kc selects K-chunk 0/1 (of 64).
__device__ __forceinline__ v16bf load_a_frag(const unsigned short* tile, int lane, int kc) {
  const int m = lane & 15, half = lane >> 4;
  const unsigned short* r = tile + m * D_DIM + kc * 32;
  v16bf a;
#pragma unroll
  for (int i = 0; i < 16; ++i) {
    const int k = i + 16 * half;
    a[i] = bfc(r[k]);
  }
  return a;
}

// B fragment (32x16 bf16, KxN) built from W stored [out=64][in=64] (row-major,
// nn.Linear convention): B[k][n] = W[nt*16+n][kc*32 + k]. lane<16: N=lane,
// K=0..15 of chunk; lane>=16: N=lane-16, K=16..31 of chunk.
__device__ __forceinline__ v16bf load_b_frag(const unsigned short* w, int lane, int nt, int kc) {
  const int n = lane & 15, half = lane >> 4;
  const unsigned short* r = w + (nt * 16 + n) * D_DIM + kc * 32 + half * 16;
  v16bf b;
#pragma unroll
  for (int i = 0; i < 16; ++i) b[i] = bfc(r[i]);
  return b;
}

// ---------- kernel 1: q = eeg@Wq^T+bq ; k = fnirs@Wk^T+bk ; score = SCALE*sum(q*k) ----------
__global__ __launch_bounds__(256) void qk_score_kernel(
    const float* __restrict__ eeg, const float* __restrict__ fnirs,
    const float* __restrict__ Wq, const float* __restrict__ bq,
    const float* __restrict__ Wk, const float* __restrict__ bk,
    float* __restrict__ score) {
  __shared__ unsigned short sE[128 * D_DIM];
  __shared__ unsigned short sF[128 * D_DIM];
  __shared__ unsigned short sWq[D_DIM * D_DIM];
  __shared__ unsigned short sWk[D_DIM * D_DIM];

  const int tid = threadIdx.x;
  const size_t rowBlock = (size_t)blockIdx.x * 128;

  // Stage 128x64 activation tiles (f32 -> bf16), coalesced float4 loads.
  const float4* gE = (const float4*)(eeg + rowBlock * D_DIM);
  const float4* gF = (const float4*)(fnirs + rowBlock * D_DIM);
#pragma unroll
  for (int i = 0; i < 8; ++i) {
    const int f4 = tid + i * 256;
    const int o = f4 * 4;
    float4 x = gE[f4];
    sE[o+0]=f2bf(x.x); sE[o+1]=f2bf(x.y); sE[o+2]=f2bf(x.z); sE[o+3]=f2bf(x.w);
    float4 y = gF[f4];
    sF[o+0]=f2bf(y.x); sF[o+1]=f2bf(y.y); sF[o+2]=f2bf(y.z); sF[o+3]=f2bf(y.w);
  }
  // Stage weights (64x64 each).
  const float4* gWq = (const float4*)Wq;
  const float4* gWk = (const float4*)Wk;
#pragma unroll
  for (int i = 0; i < 4; ++i) {
    const int f4 = tid + i * 256;
    const int o = f4 * 4;
    float4 x = gWq[f4];
    sWq[o+0]=f2bf(x.x); sWq[o+1]=f2bf(x.y); sWq[o+2]=f2bf(x.z); sWq[o+3]=f2bf(x.w);
    float4 y = gWk[f4];
    sWk[o+0]=f2bf(y.x); sWk[o+1]=f2bf(y.y); sWk[o+2]=f2bf(y.z); sWk[o+3]=f2bf(y.w);
  }
  __syncthreads();

  const int wave = tid >> 5, lane = tid & 31;
  const int half = lane >> 4, n = lane & 15;
  const int r0 = (int)rowBlock + wave * 16;

  const v16bf aE0 = load_a_frag(sE + wave * 16 * D_DIM, lane, 0);
  const v16bf aE1 = load_a_frag(sE + wave * 16 * D_DIM, lane, 1);
  const v16bf aF0 = load_a_frag(sF + wave * 16 * D_DIM, lane, 0);
  const v16bf aF1 = load_a_frag(sF + wave * 16 * D_DIM, lane, 1);

  float p[8];
#pragma unroll
  for (int j = 0; j < 8; ++j) p[j] = 0.0f;

#pragma unroll
  for (int nt = 0; nt < 4; ++nt) {
    const v16bf fq0 = load_b_frag(sWq, lane, nt, 0);
    const v16bf fq1 = load_b_frag(sWq, lane, nt, 1);
    const v16bf fk0 = load_b_frag(sWk, lane, nt, 0);
    const v16bf fk1 = load_b_frag(sWk, lane, nt, 1);
    v8f cq = {};
    cq = __builtin_amdgcn_wmma_f32_16x16x32_bf16(false, aE0, false, fq0, (short)0, cq, false, false);
    cq = __builtin_amdgcn_wmma_f32_16x16x32_bf16(false, aE1, false, fq1, (short)0, cq, false, false);
    v8f ck = {};
    ck = __builtin_amdgcn_wmma_f32_16x16x32_bf16(false, aF0, false, fk0, (short)0, ck, false, false);
    ck = __builtin_amdgcn_wmma_f32_16x16x32_bf16(false, aF1, false, fk1, (short)0, ck, false, false);
    const float bqv = bq[nt * 16 + n];
    const float bkv = bk[nt * 16 + n];
#pragma unroll
    for (int j = 0; j < 8; ++j) p[j] += (cq[j] + bqv) * (ck[j] + bkv);
  }

  // Reduce over N (16 lanes per half-wave). C layout: lane L, VGPR j -> M = j + 8*(L>=16).
#pragma unroll
  for (int j = 0; j < 8; ++j) {
    float s = p[j];
#pragma unroll
    for (int m = 1; m < 16; m <<= 1) s += __shfl_xor(s, m, 16);
    p[j] = s;
  }
  if (n == 0) {
    const int rb = r0 + half * 8;
#pragma unroll
    for (int j = 0; j < 8; ++j) score[rb + j] = p[j] * 0.125f;  // SCALE = 64^-0.5
  }
}

// ---------- kernel 2: softmax over T (per batch column b) ----------
__global__ __launch_bounds__(256) void softmax_kernel(const float* __restrict__ score,
                                                      float* __restrict__ attn) {
  const int b = blockIdx.x, tid = threadIdx.x;
  float vals[16];
  float mx = -3.4e38f;
#pragma unroll
  for (int i = 0; i < 16; ++i) {
    const int t = tid + i * 256;
    const float s = score[(size_t)t * B_DIM + b];
    vals[i] = s;
    mx = fmaxf(mx, s);
  }
#pragma unroll
  for (int m = 1; m < 32; m <<= 1) mx = fmaxf(mx, __shfl_xor(mx, m, 32));
  __shared__ float redm[8];
  if ((tid & 31) == 0) redm[tid >> 5] = mx;
  __syncthreads();
  float bm = redm[0];
#pragma unroll
  for (int w = 1; w < 8; ++w) bm = fmaxf(bm, redm[w]);

  float sum = 0.0f;
#pragma unroll
  for (int i = 0; i < 16; ++i) { vals[i] = __expf(vals[i] - bm); sum += vals[i]; }
#pragma unroll
  for (int m = 1; m < 32; m <<= 1) sum += __shfl_xor(sum, m, 32);
  __shared__ float reds[8];
  if ((tid & 31) == 0) reds[tid >> 5] = sum;
  __syncthreads();
  float bs = 0.0f;
#pragma unroll
  for (int w = 0; w < 8; ++w) bs += reds[w];
  const float inv = 1.0f / bs;
#pragma unroll
  for (int i = 0; i < 16; ++i)
    attn[(size_t)(tid + i * 256) * B_DIM + b] = vals[i] * inv;
}

// ---------- kernel 3: v = fnirs@Wv^T+bv ; out = eeg + attn*v ----------
__global__ __launch_bounds__(256) void out_kernel(
    const float* __restrict__ eeg, const float* __restrict__ fnirs,
    const float* __restrict__ Wv, const float* __restrict__ bv,
    const float* __restrict__ attn, float* __restrict__ out) {
  __shared__ unsigned short sF[128 * D_DIM];
  __shared__ unsigned short sWv[D_DIM * D_DIM];

  const int tid = threadIdx.x;
  const size_t rowBlock = (size_t)blockIdx.x * 128;

  const float4* gF = (const float4*)(fnirs + rowBlock * D_DIM);
#pragma unroll
  for (int i = 0; i < 8; ++i) {
    const int f4 = tid + i * 256;
    const int o = f4 * 4;
    float4 y = gF[f4];
    sF[o+0]=f2bf(y.x); sF[o+1]=f2bf(y.y); sF[o+2]=f2bf(y.z); sF[o+3]=f2bf(y.w);
  }
  const float4* gW = (const float4*)Wv;
#pragma unroll
  for (int i = 0; i < 4; ++i) {
    const int f4 = tid + i * 256;
    const int o = f4 * 4;
    float4 x = gW[f4];
    sWv[o+0]=f2bf(x.x); sWv[o+1]=f2bf(x.y); sWv[o+2]=f2bf(x.z); sWv[o+3]=f2bf(x.w);
  }
  __syncthreads();

  const int wave = tid >> 5, lane = tid & 31;
  const int half = lane >> 4, n = lane & 15;
  const int r0 = (int)rowBlock + wave * 16;

  const v16bf a0 = load_a_frag(sF + wave * 16 * D_DIM, lane, 0);
  const v16bf a1 = load_a_frag(sF + wave * 16 * D_DIM, lane, 1);

  float att[8];
#pragma unroll
  for (int j = 0; j < 8; ++j) att[j] = attn[r0 + j + 8 * half];

#pragma unroll
  for (int nt = 0; nt < 4; ++nt) {
    const v16bf b0 = load_b_frag(sWv, lane, nt, 0);
    const v16bf b1 = load_b_frag(sWv, lane, nt, 1);
    v8f cv = {};
    cv = __builtin_amdgcn_wmma_f32_16x16x32_bf16(false, a0, false, b0, (short)0, cv, false, false);
    cv = __builtin_amdgcn_wmma_f32_16x16x32_bf16(false, a1, false, b1, (short)0, cv, false, false);
    const float bvv = bv[nt * 16 + n];
    const int col = nt * 16 + n;
#pragma unroll
    for (int j = 0; j < 8; ++j) {
      const int row = r0 + j + 8 * half;
      const size_t idx = (size_t)row * D_DIM + col;
      out[idx] = eeg[idx] + att[j] * (cv[j] + bvv);
    }
  }
}

// ---------- launcher ----------
extern "C" void kernel_launch(void* const* d_in, const int* in_sizes, int n_in,
                              void* d_out, int out_size, void* d_ws, size_t ws_size,
                              hipStream_t stream) {
  const float* eeg   = (const float*)d_in[0];
  const float* fnirs = (const float*)d_in[1];
  const float* Wq    = (const float*)d_in[2];
  const float* bq    = (const float*)d_in[3];
  const float* Wk    = (const float*)d_in[4];
  const float* bk    = (const float*)d_in[5];
  const float* Wv    = (const float*)d_in[6];
  const float* bv    = (const float*)d_in[7];
  float* out = (float*)d_out;

  float* score = (float*)d_ws;          // ROWS floats (1 MiB)
  float* attn  = score + ROWS;          // ROWS floats (1 MiB)

  const dim3 blk(256);
  qk_score_kernel<<<ROWS / 128, blk, 0, stream>>>(eeg, fnirs, Wq, bq, Wk, bk, score);
  softmax_kernel<<<B_DIM, blk, 0, stream>>>(score, attn);
  out_kernel<<<ROWS / 128, blk, 0, stream>>>(eeg, fnirs, Wv, bv, attn, out);
}
